// OSDecoder_55018531062645
// MI455X (gfx1250) — matrix operand order repaired and checked by reading, not compiled, not run on hardware
//
#include <hip/hip_runtime.h>
#include <stdint.h>

typedef __attribute__((ext_vector_type(16))) _Float16 v16h;
typedef __attribute__((ext_vector_type(8)))  _Float16 v8h;
typedef __attribute__((ext_vector_type(8)))  float    v8f;

#define LLR_MAX_F 100.0f
static constexpr int NN    = 128;   // code length
static constexpr int KK    = 64;    // code dimension
static constexpr int BLOCK = 128;   // 4 wave32 per workgroup
static constexpr int PADH  = 8;     // f16 row padding: stride 72 halves = 36 dwords,
                                    // gcd(36,64)=4 -> 16 fragment lanes hit 16 distinct banks

struct SMem {
  float    key[NN];            // sort keys, then llr_sort
  int      idx[NN];            // idx_sort
  uint32_t rows[KK][4];        // bit-packed generator rows (sorted column order)
  int      pivot[KK];
  int      mark[NN];
  int      idx_mrb[NN];
  float    llr_mrb[NN];
  int      idx_total[NN];
  float    w[NN];              // flip gains: llr_mrb * (1 - 2c)
  unsigned long long Prow[KK]; // parity part P (64x64 bits), one u64 per row
  unsigned long long xred[KK];
  unsigned long long c_head, c_par;
  float    best_delta;
  int      best_i, best_j;
  alignas(16) _Float16 Amat[KK][KK + PADH]; // A[m][k] = P[m][k] (exact 0/1 in f16)
  alignas(16) _Float16 Bhi[KK][KK + PADH];  // TRANSPOSED: Bhi[n][k] = w_hi[k]*P[n][k]
  alignas(16) _Float16 Blo[KK][KK + PADH];  // TRANSPOSED: Blo[n][k] = w_lo[k]*P[n][k]
  float    Cmat[KK][KK];       // C = P diag(w_par) P^T  (f32 accum)
  float    s[KK];              // single-flip gains
  float    r_delta[BLOCK];
  int      r_rank[BLOCK];
};

__device__ __forceinline__ v16h frag_cat(v8h lo, v8h hi) {
  return __builtin_shufflevector(lo, hi, 0, 1, 2, 3, 4, 5, 6, 7,
                                         8, 9, 10, 11, 12, 13, 14, 15);
}

__global__ __launch_bounds__(BLOCK)
void osd_decode_kernel(const float* __restrict__ llrs,
                       const float* __restrict__ gm,
                       const int*   __restrict__ t_ptr,
                       float*       __restrict__ out) {
  __shared__ SMem sm;
  const int b   = blockIdx.x;
  const int tid = threadIdx.x;
  const int t_order = t_ptr[0];

  // ---- load + clip, init sort keys (key = -|llr|, tie-break by index => stable) ----
  {
    float v = llrs[b * NN + tid];
    v = fminf(fmaxf(v, -LLR_MAX_F), LLR_MAX_F);
    sm.key[tid] = -fabsf(v);
    sm.idx[tid] = tid;
  }
  __syncthreads();

  // ---- bitonic sort of 128 (key, idx) pairs, ascending lexicographic ----
  for (int size = 2; size <= NN; size <<= 1) {
    for (int stride = size >> 1; stride > 0; stride >>= 1) {
      if (tid < NN / 2) {
        int lo = (tid / stride) * (stride * 2) + (tid % stride);
        int hi = lo + stride;
        bool up = ((lo & size) == 0);
        float k0 = sm.key[lo], k1 = sm.key[hi];
        int   i0 = sm.idx[lo], i1 = sm.idx[hi];
        bool swp = up ? ((k0 > k1) || (k0 == k1 && i0 > i1))
                      : ((k0 < k1) || (k0 == k1 && i0 < i1));
        if (swp) {
          sm.key[lo] = k1; sm.key[hi] = k0;
          sm.idx[lo] = i1; sm.idx[hi] = i0;
        }
      }
      __syncthreads();
    }
  }

  // ---- llr_sort (reuse key array) ----
  {
    float v = llrs[b * NN + sm.idx[tid]];
    sm.key[tid] = fminf(fmaxf(v, -LLR_MAX_F), LLR_MAX_F);
  }

  // ---- pack permuted generator rows into bits ----
  if (tid < KK) {
    uint32_t wbits[4] = {0u, 0u, 0u, 0u};
    const float* grow = gm + tid * NN;
    for (int j = 0; j < NN; ++j) {
      if (grow[sm.idx[j]] > 0.5f) wbits[j >> 5] |= (1u << (j & 31));
    }
    for (int q = 0; q < 4; ++q) sm.rows[tid][q] = wbits[q];
  }

  // ---- GF(2) elimination with first-set-bit pivoting (== argmax on 0/1 floats) ----
  for (int i = 0; i < KK; ++i) {
    __syncthreads();
    if (tid == 0) {
      int p = 0;
      for (int q = 0; q < 4; ++q) {
        uint32_t v = sm.rows[i][q];
        if (v) { p = q * 32 + (__ffs(v) - 1); break; }
      }
      sm.pivot[i] = p;
    }
    __syncthreads();
    int p = sm.pivot[i];
    if (tid < KK && tid != i) {
      if ((sm.rows[tid][p >> 5] >> (p & 31)) & 1u) {
        for (int q = 0; q < 4; ++q) sm.rows[tid][q] ^= sm.rows[i][q];
      }
    }
  }
  __syncthreads();

  // ---- mark pivot columns, build idx_mrb = [pivots | non-pivots ascending] ----
  sm.mark[tid] = 0;
  __syncthreads();
  if (tid < KK) sm.mark[sm.pivot[tid]] = 1;
  __syncthreads();
  if (tid < KK) sm.idx_mrb[tid] = sm.pivot[tid];
  if (sm.mark[tid] == 0) {
    int rank = 0;
    for (int j = 0; j < tid; ++j) rank += (sm.mark[j] == 0);
    sm.idx_mrb[KK + rank] = tid;
  }
  __syncthreads();

  // ---- llr_mrb, idx_total ----
  {
    int jm = sm.idx_mrb[tid];
    sm.llr_mrb[tid]   = sm.key[jm];
    sm.idx_total[tid] = sm.idx[jm];
  }
  __syncthreads();

  // ---- extract parity part P (columns idx_mrb[64..127]) ----
  if (tid < KK) {
    unsigned long long pr = 0ull;
    for (int t = 0; t < KK; ++t) {
      int col = sm.idx_mrb[KK + t];
      if ((sm.rows[tid][col >> 5] >> (col & 31)) & 1u) pr |= (1ull << t);
    }
    sm.Prow[tid] = pr;
  }
  __syncthreads();

  // ---- base codeword c = [u_hat | u_hat @ P] via XOR reductions ----
  bool uhat = (tid < KK) ? (sm.llr_mrb[tid] > 0.0f) : false;
  if (tid < KK) sm.xred[tid] = uhat ? sm.Prow[tid] : 0ull;
  __syncthreads();
  for (int off = 32; off > 0; off >>= 1) {
    if (tid < off) sm.xred[tid] ^= sm.xred[tid + off];
    __syncthreads();
  }
  if (tid == 0) sm.c_par = sm.xred[0];
  __syncthreads();
  if (tid < KK) sm.xred[tid] = uhat ? (1ull << tid) : 0ull;
  __syncthreads();
  for (int off = 32; off > 0; off >>= 1) {
    if (tid < off) sm.xred[tid] ^= sm.xred[tid + off];
    __syncthreads();
  }
  if (tid == 0) sm.c_head = sm.xred[0];
  __syncthreads();

  // ---- flip gains w_n = llr_mrb_n * (1 - 2 c_n) ----
  {
    int cbit = (tid < KK) ? (int)((sm.c_head >> tid) & 1ull)
                          : (int)((sm.c_par >> (tid - KK)) & 1ull);
    sm.w[tid] = sm.llr_mrb[tid] * (1.0f - 2.0f * (float)cbit);
  }
  __syncthreads();

  // ---- build WMMA operands: A = P; Bt{hi,lo}[n][k] = w_par{hi,lo}[k]*P[n][k] ----
  for (int e = tid; e < KK * KK; e += BLOCK) {
    int m = e >> 6, k = e & 63;
    sm.Amat[m][k] = (_Float16)(float)((sm.Prow[m] >> k) & 1ull);
  }
  for (int e = tid; e < KK * KK; e += BLOCK) {
    int n = e >> 6, k = e & 63;
    float wpar = sm.w[KK + k];
    _Float16 whi = (_Float16)wpar;
    _Float16 wlo = (_Float16)(wpar - (float)whi);
    int bit = (int)((sm.Prow[n] >> k) & 1ull);
    sm.Bhi[n][k] = bit ? whi : (_Float16)0.0f;
    sm.Blo[n][k] = bit ? wlo : (_Float16)0.0f;
  }
  __syncthreads();

  // ---- C = P diag(w_par) P^T : 4x4 tiles of v_wmma_f32_16x16x32_f16 ----
  // Fragment loads are contiguous 16B ds_load_b128 chunks (bank-conflict-free via PADH).
  {
    const int wave  = tid >> 5;
    const int lane  = tid & 31;
    const int lrow  = lane & 15;
    const int lhalf = lane >> 4;
    const int mtile = wave;                       // each wave owns 16 rows of C
    const int mrow  = mtile * 16 + lrow;

    // A fragments for both K-steps (reused across ntile and both precision passes).
    // A 16x32 f16 layout: lane half selects K groups {0-7,16-23}/{8-15,24-31}.
    v16h afrag[2];
#pragma unroll
    for (int ks = 0; ks < 2; ++ks) {
      int ko = ks * 32;
      v8h a0 = *(const v8h*)&sm.Amat[mrow][ko + 8 * lhalf];
      v8h a1 = *(const v8h*)&sm.Amat[mrow][ko + 16 + 8 * lhalf];
      afrag[ks] = frag_cat(a0, a1);
    }

    for (int ntile = 0; ntile < 4; ++ntile) {
      const int ncol = ntile * 16 + lrow;
      v8f acc = {};
#pragma unroll
      for (int pass = 0; pass < 2; ++pass) {
        const _Float16 (*Bt)[KK + PADH] = (pass == 0) ? sm.Bhi : sm.Blo;
#pragma unroll
        for (int ks = 0; ks < 2; ++ks) {
          int ko = ks * 32;
          // B 32x16 f16 layout: lane half selects K groups {0-7,16-23}/{8-15,24-31} (same as A);
          // with Bt[n][k] these are two 8-half chunks.
          v8h b0 = *(const v8h*)&Bt[ncol][ko + 8 * lhalf];
          v8h b1 = *(const v8h*)&Bt[ncol][ko + 16 + 8 * lhalf];
          v16h bvec = frag_cat(b0, b1);
          acc = __builtin_amdgcn_wmma_f32_16x16x32_f16(
              /*neg_a=*/false, afrag[ks], /*neg_b=*/false, bvec,
              /*c_mod=*/(short)0, acc, /*reuse_a=*/false, /*reuse_b=*/false);
        }
      }
#pragma unroll
      for (int r = 0; r < 8; ++r) {
        int m = mtile * 16 + r + 8 * lhalf;       // C/D layout: VGPR r -> M = r + 8*half
        sm.Cmat[m][ntile * 16 + lrow] = acc[r];
      }
    }
  }
  __syncthreads();

  // ---- single-flip gains from GEMM diagonal: s_i = C_ii + w_i ----
  if (tid < KK) sm.s[tid] = sm.Cmat[tid][tid] + sm.w[tid];
  __syncthreads();

  // ---- order-1 search (strict improvement over base, first index on ties) ----
  if (tid == 0) {
    float bd = 0.0f; int bi = -1;
    if (t_order >= 1) {
      for (int i = 0; i < KK; ++i) {
        float d = sm.s[i];
        if (d > bd) { bd = d; bi = i; }
      }
    }
    sm.best_delta = bd; sm.best_i = bi; sm.best_j = -1;
  }
  __syncthreads();

  // ---- order-2 search: delta_ij = s_i + s_j - 2*C_ij over 2016 pairs ----
  {
    float local_d = -__builtin_huge_valf();
    int   local_r = 0x7fffffff;
    int r = 0;
    for (int i = 0; i < KK - 1; ++i) {
      float si = sm.s[i];
      for (int j = i + 1; j < KK; ++j, ++r) {
        if ((r & (BLOCK - 1)) == tid) {
          float d = si + sm.s[j] - 2.0f * sm.Cmat[i][j];
          if (d > local_d || (d == local_d && r < local_r)) { local_d = d; local_r = r; }
        }
      }
    }
    sm.r_delta[tid] = local_d;
    sm.r_rank[tid]  = local_r;
  }
  __syncthreads();
  for (int off = BLOCK / 2; off > 0; off >>= 1) {
    if (tid < off) {
      float d2 = sm.r_delta[tid + off]; int r2 = sm.r_rank[tid + off];
      if (d2 > sm.r_delta[tid] || (d2 == sm.r_delta[tid] && r2 < sm.r_rank[tid])) {
        sm.r_delta[tid] = d2; sm.r_rank[tid] = r2;
      }
    }
    __syncthreads();
  }
  if (tid == 0) {
    if (t_order >= 2 && sm.r_delta[0] > sm.best_delta) {
      int rr = sm.r_rank[0];
      int i = 0;
      while (rr >= (KK - 1 - i)) { rr -= (KK - 1 - i); ++i; }
      sm.best_i = i;
      sm.best_j = i + 1 + rr;
      sm.best_delta = sm.r_delta[0];
    }
  }
  __syncthreads();

  // ---- apply winning flips, un-permute to original column order ----
  {
    unsigned long long ch = sm.c_head, cp = sm.c_par;
    int bi = sm.best_i, bj = sm.best_j;
    if (bi >= 0) { ch ^= (1ull << bi); cp ^= sm.Prow[bi]; }
    if (bj >= 0) { ch ^= (1ull << bj); cp ^= sm.Prow[bj]; }
    int bit = (tid < KK) ? (int)((ch >> tid) & 1ull)
                         : (int)((cp >> (tid - KK)) & 1ull);
    out[b * NN + sm.idx_total[tid]] = (float)bit;
  }
}

extern "C" void kernel_launch(void* const* d_in, const int* in_sizes, int n_in,
                              void* d_out, int out_size, void* d_ws, size_t ws_size,
                              hipStream_t stream) {
  const float* llrs = (const float*)d_in[0];   // (bs, 128) float32
  const float* gmp  = (const float*)d_in[1];   // (64, 128) float32
  const int*   t    = (const int*)d_in[2];     // scalar order
  float* out = (float*)d_out;                  // (bs, 128) float32
  (void)d_ws; (void)ws_size; (void)n_in; (void)out_size;
  int bs = in_sizes[0] / NN;
  osd_decode_kernel<<<dim3(bs), dim3(BLOCK), 0, stream>>>(llrs, gmp, t, out);
}
